// FDEFractalNoPropNetwork_41025527611630
// MI455X (gfx1250) — matrix operand order repaired and checked by reading, not compiled, not run on hardware
//
#include <hip/hip_runtime.h>
#include <math.h>

// ---------------------------------------------------------------------------
// FDE "NoProp" network: 23 GEMMs with fused epilogues on CDNA5 bf16 WMMA
// (fp32 accumulate).  B=2048, IN=1024, H=2048, OUT=1024, T=10.
// Double-buffered LDS pipeline, BK=64 (2 WMMA K-steps per stage).
// ---------------------------------------------------------------------------

typedef __attribute__((ext_vector_type(16))) __bf16 v16bf;
typedef __attribute__((ext_vector_type(8)))  float  v8f;

constexpr int BM = 128;       // workgroup M tile
constexpr int BN = 128;       // workgroup N tile
constexpr int BK = 64;        // K per stage = 2 x WMMA-K(32)
constexpr int LDA  = BK + 8;  // padded bf16 row stride (144B, 16B aligned)
constexpr int LDBT = BK + 8;

struct __align__(16) U4 { unsigned int x, y, z, w; };

union Frag {
    v16bf bf;
    U4    u[2];
};

// Epilogue modes
//  0: C = acc + bias
//  1: C = silu(acc + bias)
//  2: C = alpha*(acc+bias) + (1-alpha)*zprev + nscale*noise
__global__ __launch_bounds__(256, 2)
void fde_gemm_bf16(const float* __restrict__ A1, const float* __restrict__ A2,
                   const float* __restrict__ W,  const float* __restrict__ bias,
                   const float* __restrict__ zprev, const float* __restrict__ nois,
                   float* __restrict__ C,
                   int M, int N, int K1, int K2,
                   float alpha, float nscale, int mode)
{
    __shared__ __bf16 sA[2][BM * LDA];   // A tile, row-major (m, k)
    __shared__ __bf16 sB[2][BN * LDBT];  // W tile, transposed: sB[n*LDBT + k]

    const int tid   = threadIdx.x;
    const int lane  = tid & 31;
    const int wave  = tid >> 5;
    const int wm    = wave & 3;    // 4 waves along M, 32 rows each
    const int wn    = wave >> 2;   // 2 waves along N, 64 cols each
    const int l15   = lane & 15;
    const int lhalf = lane >> 4;

    const int m0 = blockIdx.y * BM;
    const int n0 = blockIdx.x * BN;
    const int Ktot = K1 + K2;

    v8f acc[2][4] = {};

    // ---- cooperative stage of one BKx(BM|BN) tile pair into LDS buffer `buf` ----
    auto stage = [&](int buf, int k0) {
        // Select A segment (handles concat([z, x_embed]) without materializing it)
        const float* Ap; int ka, lda;
        if (k0 < K1) { Ap = A1; ka = k0;      lda = K1; }
        else         { Ap = A2; ka = k0 - K1; lda = K2; }

        // A tile: BM x BK fp32 -> bf16 (2048 float4s, 8 per thread)
        #pragma unroll
        for (int it = 0; it < 8; ++it) {
            int q   = tid + it * 256;      // 0..2047
            int row = q >> 4;              // 0..127
            int col = (q & 15) << 2;       // 0..60
            float4 f = *(const float4*)(Ap + (size_t)(m0 + row) * lda + ka + col);
            __bf16* d = &sA[buf][row * LDA + col];
            d[0] = (__bf16)f.x; d[1] = (__bf16)f.y;
            d[2] = (__bf16)f.z; d[3] = (__bf16)f.w;
        }
        // W tile: BK x BN fp32 -> transposed bf16 (2048 float4s, 8 per thread)
        #pragma unroll
        for (int it = 0; it < 8; ++it) {
            int q = tid + it * 256;        // 0..2047
            int k = q >> 5;                // 0..63
            int n = (q & 31) << 2;         // 0..124
            float4 f = *(const float4*)(W + (size_t)(k0 + k) * N + n0 + n);
            sB[buf][(n + 0) * LDBT + k] = (__bf16)f.x;
            sB[buf][(n + 1) * LDBT + k] = (__bf16)f.y;
            sB[buf][(n + 2) * LDBT + k] = (__bf16)f.z;
            sB[buf][(n + 3) * LDBT + k] = (__bf16)f.w;
        }
    };

    // ---- software pipeline: prologue fill, then overlap stage(next) w/ WMMA ----
    stage(0, 0);
    int cur = 0;
    for (int k0 = 0; k0 < Ktot; k0 += BK) {
        __syncthreads();   // buf `cur` visible to all; buf `cur^1` free to fill

        // warm GL2 two stages ahead (global_prefetch_b8)
        if (k0 + 2 * BK < Ktot)
            __builtin_prefetch(W + (size_t)(k0 + 2 * BK) * N + n0 + (lane << 2), 0, 1);

        // issue next tile's global loads early so they hide under the WMMAs
        if (k0 + BK < Ktot)
            stage(cur ^ 1, k0 + BK);

        // 2 K-steps x 8 WMMAs on the current buffer
        #pragma unroll
        for (int kk = 0; kk < BK; kk += 32) {
            Frag a[2], b[4];
            #pragma unroll
            for (int mt = 0; mt < 2; ++mt) {
                // A 16x32 bf16: lane<16 -> K 0..7 & 16..23 ; lane>=16 -> K 8..15 & 24..31
                int row = wm * 32 + mt * 16 + l15;
                const __bf16* p = &sA[cur][row * LDA + kk + lhalf * 8];
                a[mt].u[0] = *(const U4*)(p);
                a[mt].u[1] = *(const U4*)(p + 16);
            }
            #pragma unroll
            for (int nt = 0; nt < 4; ++nt) {
                // B 32x16 bf16: lane<16 -> K 0..7 & 16..23 of col N ; lane>=16 -> K 8..15 & 24..31
                int col = wn * 64 + nt * 16 + l15;
                const __bf16* p = &sB[cur][col * LDBT + kk + lhalf * 8];
                b[nt].u[0] = *(const U4*)(p);
                b[nt].u[1] = *(const U4*)(p + 16);
            }
            #pragma unroll
            for (int mt = 0; mt < 2; ++mt)
                #pragma unroll
                for (int nt = 0; nt < 4; ++nt)
                    acc[mt][nt] = __builtin_amdgcn_wmma_f32_16x16x32_bf16(
                        false, a[mt].bf, false, b[nt].bf,
                        (short)0, acc[mt][nt], false, false);
        }
        cur ^= 1;
    }

    // ---- fused epilogue + store ----
    #pragma unroll
    for (int mt = 0; mt < 2; ++mt) {
        #pragma unroll
        for (int nt = 0; nt < 4; ++nt) {
            const int gN  = n0 + wn * 64 + nt * 16 + l15;
            const int gM0 = m0 + wm * 32 + mt * 16 + lhalf * 8;
            const float bval = bias[gN];
            #pragma unroll
            for (int r = 0; r < 8; ++r) {
                const int gM = gM0 + r;
                float v = acc[mt][nt][r] + bval;
                if (mode == 1) {
                    v = v * (1.0f / (1.0f + __expf(-v)));       // SiLU
                } else if (mode == 2) {
                    const size_t idx = (size_t)gM * N + gN;
                    v = alpha * v + (1.0f - alpha) * zprev[idx] + nscale * nois[idx];
                }
                C[(size_t)gM * N + gN] = v;
            }
        }
    }
}

// ---------------------------------------------------------------------------
// Host side: 23 GEMM launches chained on `stream`.
// ---------------------------------------------------------------------------
extern "C" void kernel_launch(void* const* d_in, const int* in_sizes, int n_in,
                              void* d_out, int out_size, void* d_ws, size_t ws_size,
                              hipStream_t stream) {
    const float* x     = (const float*)d_in[0];
    const float* z0    = (const float*)d_in[1];
    const float* noise = (const float*)d_in[2];
    const float* w1_in = (const float*)d_in[3];
    const float* b1_in = (const float*)d_in[4];
    const float* w2_in = (const float*)d_in[5];
    const float* b2_in = (const float*)d_in[6];
    const float* wA    = (const float*)d_in[7];
    const float* bA    = (const float*)d_in[8];
    const float* wB    = (const float*)d_in[9];
    const float* bB    = (const float*)d_in[10];
    const float* wC    = (const float*)d_in[11];
    const float* bC    = (const float*)d_in[12];
    float* out = (float*)d_out;

    const int Bn = 2048, IN = 1024, H = 2048, OUT = 1024, T = 10;
    const size_t BH = (size_t)Bn * H;

    float* ws     = (float*)d_ws;
    float* buf_u  = ws;            // B*H scratch: h, then u per step
    float* buf_xe = ws + BH;       // x_embed
    float* zA     = ws + 2 * BH;   // z ping
    float* zB     = ws + 3 * BH;   // z pong

    const dim3 block(256);
    auto run = [&](const float* A1, const float* A2, int K1, int K2,
                   const float* W, const float* bias, float* C,
                   int M, int N, int mode,
                   const float* zp, const float* np, float a, float ns) {
        dim3 grid((unsigned)(N / BN), (unsigned)(M / BM));
        fde_gemm_bf16<<<grid, block, 0, stream>>>(A1, A2, W, bias, zp, np, C,
                                                  M, N, K1, K2, a, ns, mode);
    };

    // input embedding
    run(x,     nullptr, IN, 0, w1_in, b1_in, buf_u,  Bn, H, 1, nullptr, nullptr, 0.f, 0.f);
    run(buf_u, nullptr, H,  0, w2_in, b2_in, buf_xe, Bn, H, 0, nullptr, nullptr, 0.f, 0.f);

    // T diffusion-style blocks
    const float* zc = z0;
    float* zn = zA;
    for (int t = 0; t < T; ++t) {
        const float a_t = 0.99f - 0.01f * (float)t;         // linspace(0.99, 0.9, 10)
        const float ns  = sqrtf(1.0f - a_t);
        // u = silu(concat([z, x_embed]) @ wA_t + bA_t)   (K = 2H, split sources)
        run(zc, buf_xe, H, H, wA + (size_t)t * 2 * H * H, bA + (size_t)t * H,
            buf_u, Bn, H, 1, nullptr, nullptr, 0.f, 0.f);
        // z = a*(u @ wB_t + bB_t) + (1-a)*z + ns*noise_t
        run(buf_u, nullptr, H, 0, wB + (size_t)t * H * H, bB + (size_t)t * H,
            zn, Bn, H, 2, zc, noise + (size_t)t * BH, a_t, ns);
        zc = zn;
        zn = (zn == zA) ? zB : zA;
    }

    // classifier
    run(zc, nullptr, H, 0, wC, bC, out, Bn, OUT, 0, nullptr, nullptr, 0.f, 0.f);
}
